// BayesianSNN_60473139528335
// MI455X (gfx1250) — matrix operand order repaired and checked by reading, not compiled, not run on hardware
//
#include <hip/hip_runtime.h>

// ---------------------------------------------------------------------------
// BayesianSNN on gfx1250 (MI455X):
//   prep:  w = mu + eps*exp(0.5*logvar) -> bf16, stored TRANSPOSED [N][K]
//          x -> bf16 ; zero membranes + output
//   loop:  per timestep, fused GEMM+LIF via v_wmma_f32_16x16x32_bf16,
//          64x64 wave tiles, double-buffered LDS fed by
//          global_load_async_to_lds_b128 (ASYNCcnt-tracked DMA).
// ---------------------------------------------------------------------------

typedef __attribute__((ext_vector_type(16))) __bf16 v16bf;
typedef __attribute__((ext_vector_type(8)))  float  v8f;

union ABFrag { uint4 u[2]; v16bf v; };

// Problem dims (fixed by reference)
#define BATCH  2048
#define TSTEPS 5
#define D_IN   2048
#define D_H    4096
#define D_OUT  1024

#define BETA_F  0.95f
#define THR_F   1.0f

// Tiling: 128x128 block tile, 4 waves (128 threads), 64x64 per wave, BK=32.
#define BM 128
#define BN 128
#define BK 32
#define NTHREADS 128
#define LDS_PAD 40   // row stride in bf16 elements (32 data + 8 pad) = 80B rows

// CDNA5 async copy: per-lane 16B global -> LDS, tracked by ASYNCcnt.
#define ASYNC_CP16(ldsoff_u32, gptr)                                     \
  asm volatile("global_load_async_to_lds_b128 %0, %1, off"               \
               :: "v"(ldsoff_u32), "v"(gptr) : "memory")
#define WAIT_ASYNC0() asm volatile("s_wait_asynccnt 0x0" ::: "memory")

// ---------------------------------------------------------------------------
// Prep kernels
// ---------------------------------------------------------------------------

// Reparameterize and store transposed: outT[n*K + k] = bf16(w[k*N + n]).
// 32x32 tiles through LDS; both global read and write coalesced.
__global__ __launch_bounds__(256) void reparam_transpose_bf16(
    const float* __restrict__ mu, const float* __restrict__ lv,
    const float* __restrict__ eps, __bf16* __restrict__ outT,
    int K, int N) {
  __shared__ float t[32][33];
  const int n0 = blockIdx.x * 32;
  const int k0 = blockIdx.y * 32;
  const int tx = threadIdx.x;           // 0..31
  const int ty = threadIdx.y;           // 0..7
#pragma unroll
  for (int j = 0; j < 4; ++j) {
    int k = k0 + ty + j * 8;
    size_t idx = (size_t)k * N + (n0 + tx);
    float w = mu[idx] + eps[idx] * __expf(0.5f * lv[idx]);
    t[ty + j * 8][tx] = w;
  }
  __syncthreads();
#pragma unroll
  for (int j = 0; j < 4; ++j) {
    int n = n0 + ty + j * 8;
    outT[(size_t)n * K + (k0 + tx)] = (__bf16)t[tx][ty + j * 8];
  }
}

__global__ void f32_to_bf16(const float* __restrict__ in,
                            __bf16* __restrict__ out, int n) {
  int stride = gridDim.x * blockDim.x;
  for (int i = blockIdx.x * blockDim.x + threadIdx.x; i < n; i += stride)
    out[i] = (__bf16)in[i];
}

__global__ void zero_f32(float* __restrict__ p, int n) {
  int stride = gridDim.x * blockDim.x;
  for (int i = blockIdx.x * blockDim.x + threadIdx.x; i < n; i += stride)
    p[i] = 0.0f;
}

// ---------------------------------------------------------------------------
// Fused GEMM + LIF step.
//   cur = A @ B   A: [M,K] bf16 row-major (stride lda)
//                 Bt: [N,K] bf16 row-major (B transposed; stride = K)
//   mem_new = beta*mem_old + cur - (mem_old > thr)*thr ; spk = mem_new > thr
//   MODE 0: spk -> bf16 buffer (layer 1);  MODE 1: out_acc += spk (layer 2)
// ---------------------------------------------------------------------------
template <int MODE>
__global__ __launch_bounds__(NTHREADS) void gemm_lif(
    const __bf16* __restrict__ A, int lda,
    const __bf16* __restrict__ Bt,           // [N][K], row stride == K
    float* __restrict__ mem,
    __bf16* __restrict__ spk_out,
    float*  __restrict__ out_acc,
    int M, int N, int K) {

  __shared__ __bf16 lA[2][BM * LDS_PAD];     // [m][k]
  __shared__ __bf16 lB[2][BN * LDS_PAD];     // [n][k]

  const int tid  = threadIdx.x;
  const int lane = tid & 31;
  const int wv   = tid >> 5;                 // 0..3
  const int wm   = (wv >> 1) * 64;           // wave M origin in block tile
  const int wn   = (wv & 1) * 64;            // wave N origin in block tile
  const int lm   = lane & 15;
  const int lh   = lane >> 4;                // 0 or 1

  const int blockN = blockIdx.x * BN;
  const int blockM = blockIdx.y * BM;

  // ---- per-thread staging: 4 A chunks + 4 B chunks of 8 bf16 each ---------
  // 128x32 tile = 512 chunks of 8; 128 threads -> 4 chunks each.
  int rA[4], kA[4];
  const __bf16* gA[4];
  const __bf16* gB[4];
  uint32_t dA[4][2], dB[4][2];
#pragma unroll
  for (int c = 0; c < 4; ++c) {
    int chunk = tid + c * NTHREADS;
    rA[c] = chunk >> 2;
    kA[c] = (chunk & 3) * 8;
    gA[c] = A  + (size_t)(blockM + rA[c]) * lda + kA[c];
    gB[c] = Bt + (size_t)(blockN + rA[c]) * K   + kA[c];
#pragma unroll
    for (int b = 0; b < 2; ++b) {
      dA[c][b] = (uint32_t)(size_t)&lA[b][rA[c] * LDS_PAD + kA[c]];
      dB[c][b] = (uint32_t)(size_t)&lB[b][rA[c] * LDS_PAD + kA[c]];
    }
  }

  v8f acc[4][4] = {};                        // 16 C tiles of 16x16 f32

  // prologue: async-stage buffer 0
#pragma unroll
  for (int c = 0; c < 4; ++c) ASYNC_CP16(dA[c][0], gA[c]);
#pragma unroll
  for (int c = 0; c < 4; ++c) ASYNC_CP16(dB[c][0], gB[c]);

  for (int kt = 0; kt < K; kt += BK) {
    const int cur = (kt >> 5) & 1;

    WAIT_ASYNC0();          // our stage-cur DMA complete
    __syncthreads();        // all waves staged; all reads of buf nxt retired

    if (kt + BK < K) {      // async-stage next buffer, overlapped with MMA
      const int nxt = cur ^ 1;
#pragma unroll
      for (int c = 0; c < 4; ++c) ASYNC_CP16(dA[c][nxt], gA[c] + kt + BK);
#pragma unroll
      for (int c = 0; c < 4; ++c) ASYNC_CP16(dB[c][nxt], gB[c] + kt + BK);
    }

    // ---- fragments (documented CDNA5 wave32 WMMA layouts) ----
    ABFrag a[4];
#pragma unroll
    for (int mf = 0; mf < 4; ++mf) {
      int r = wm + mf * 16 + lm;             // lane -> row M
      a[mf].u[0] = *(const uint4*)&lA[cur][r * LDS_PAD + lh * 8];
      a[mf].u[1] = *(const uint4*)&lA[cur][r * LDS_PAD + 16 + lh * 8];
    }
#pragma unroll
    for (int nf = 0; nf < 4; ++nf) {
      int c = wn + nf * 16 + lm;             // lane -> col N
      ABFrag b;
      b.u[0] = *(const uint4*)&lB[cur][c * LDS_PAD + lh * 8];
      b.u[1] = *(const uint4*)&lB[cur][c * LDS_PAD + 16 + lh * 8];
#pragma unroll
      for (int mf = 0; mf < 4; ++mf)
        acc[mf][nf] = __builtin_amdgcn_wmma_f32_16x16x32_bf16(
            false, a[mf].v, false, b.v, (short)0, acc[mf][nf],
            false, false);
    }
  }

  // ---- LIF epilogue on C layout: VGPR r -> M = r + 8*lh, N = lane&15 ----
#pragma unroll
  for (int mf = 0; mf < 4; ++mf)
#pragma unroll
    for (int nf = 0; nf < 4; ++nf)
#pragma unroll
      for (int r = 0; r < 8; ++r) {
        int gm = blockM + wm + mf * 16 + r + lh * 8;
        int gn = blockN + wn + nf * 16 + lm;
        size_t idx = (size_t)gm * N + gn;
        float cur  = acc[mf][nf][r];
        float old  = mem[idx];
        float nm   = BETA_F * old + cur - ((old > THR_F) ? THR_F : 0.0f);
        mem[idx]   = nm;
        float spk  = (nm - THR_F) > 0.0f ? 1.0f : 0.0f;
        if (MODE == 0) {
          spk_out[idx] = (__bf16)spk;
        } else {
          out_acc[idx] += spk;
        }
      }
}

// ---------------------------------------------------------------------------
// Host-side launcher
// ---------------------------------------------------------------------------
extern "C" void kernel_launch(void* const* d_in, const int* in_sizes, int n_in,
                              void* d_out, int out_size, void* d_ws, size_t ws_size,
                              hipStream_t stream) {
  const float* x     = (const float*)d_in[0];
  const float* w1_mu = (const float*)d_in[1];
  const float* w1_lv = (const float*)d_in[2];
  const float* w2_mu = (const float*)d_in[3];
  const float* w2_lv = (const float*)d_in[4];
  const float* eps1  = (const float*)d_in[5];
  const float* eps2  = (const float*)d_in[6];
  float* out = (float*)d_out;

  // ---- carve workspace (256B aligned) ----
  char* p = (char*)d_ws;
  auto carve = [&](size_t bytes) -> void* {
    void* r = (void*)p;
    p += (bytes + 255) & ~(size_t)255;
    return r;
  };
  const size_t n_w1 = (size_t)D_IN * D_H;
  const size_t n_w2 = (size_t)D_H * D_OUT;
  const size_t n_x  = (size_t)BATCH * TSTEPS * D_IN;
  const size_t n_m1 = (size_t)BATCH * D_H;
  const size_t n_m2 = (size_t)BATCH * D_OUT;

  __bf16* w1t  = (__bf16*)carve(n_w1 * 2);   // [D_H][D_IN]
  __bf16* w2t  = (__bf16*)carve(n_w2 * 2);   // [D_OUT][D_H]
  __bf16* xb   = (__bf16*)carve(n_x * 2);
  __bf16* spk1 = (__bf16*)carve(n_m1 * 2);
  float*  mem1 = (float*)carve(n_m1 * 4);
  float*  mem2 = (float*)carve(n_m2 * 4);

  // ---- prep ----
  dim3 tb(32, 8);
  dim3 gt1(D_H / 32, D_IN / 32);             // w1: K=D_IN rows, N=D_H cols
  dim3 gt2(D_OUT / 32, D_H / 32);            // w2: K=D_H,  N=D_OUT
  reparam_transpose_bf16<<<gt1, tb, 0, stream>>>(w1_mu, w1_lv, eps1, w1t, D_IN, D_H);
  reparam_transpose_bf16<<<gt2, tb, 0, stream>>>(w2_mu, w2_lv, eps2, w2t, D_H, D_OUT);

  const int EB = 256, EG = 2048;
  f32_to_bf16<<<EG, EB, 0, stream>>>(x, xb, (int)n_x);
  zero_f32<<<EG, EB, 0, stream>>>(mem1, (int)n_m1);
  zero_f32<<<EG, EB, 0, stream>>>(mem2, (int)n_m2);
  zero_f32<<<EG, EB, 0, stream>>>(out, out_size);

  dim3 blk(NTHREADS);
  dim3 g1(D_H / BN, BATCH / BM);             // 32 x 16 = 512 blocks
  dim3 g2(D_OUT / BN, BATCH / BM);           //  8 x 16 = 128 blocks

  // ---- T sequential steps ----
  for (int t = 0; t < TSTEPS; ++t) {
    // x_t row b is at xb[(b*T + t)*D_IN] -> base offset t*D_IN, lda = T*D_IN
    gemm_lif<0><<<g1, blk, 0, stream>>>(
        xb + (size_t)t * D_IN, TSTEPS * D_IN,
        w1t,
        mem1, spk1, (float*)nullptr,
        BATCH, D_H, D_IN);

    gemm_lif<1><<<g2, blk, 0, stream>>>(
        spk1, D_H,
        w2t,
        mem2, (__bf16*)nullptr, out,
        BATCH, D_OUT, D_H);
  }
}
